// LearnableFF_85246510891766
// MI455X (gfx1250) — matrix operand order repaired and checked by reading, not compiled, not run on hardware
//
#include <hip/hip_runtime.h>
#include <hip/hip_bf16.h>

// ---------------------------------------------------------------------------
// MI455X (gfx1250, wave32) implementation of the MPNN + docking reference.
//  - All dense GEMMs:  v_wmma_f32_16x16x32_bf16 (bf16 in, f32 accum),
//    K templated + fully unrolled -> straight-line SSA, no acc/frag copies
//  - W_edge kept in bf16 (151 MB): halves traffic of the dominant tensor
//  - per-edge mat-vec message passing: wave-per-edge VALU + f32 atomics
//  - docking: one 512-thread block / graph, shuffle+LDS reductions
// ---------------------------------------------------------------------------

typedef __bf16 bf16;
typedef __attribute__((ext_vector_type(16))) __bf16 v16bf;
typedef __attribute__((ext_vector_type(8)))  __bf16 v8bf;
typedef __attribute__((ext_vector_type(2)))  __bf16 v2bf;
typedef __attribute__((ext_vector_type(8)))  float  v8f;

#define GG      16
#define NLIG    48
#define NREC    512
#define ELIG    128
#define EREC    1024
#define DINP_N  288   // 260 padded to multiple of 32
#define DINP_E  160   // 130 padded to multiple of 32
#define OPT_LRF 1e-3f

// ------------------------------- prep kernels ------------------------------

// dst[N, Kp] (bf16) = transpose(src[K, N]) with zero K-padding
__global__ __launch_bounds__(256) void wtrans_kernel(bf16* __restrict__ dst,
    const float* __restrict__ src, int K, int N, int Kp) {
  int idx = blockIdx.x * blockDim.x + threadIdx.x;
  if (idx >= N * Kp) return;
  int n = idx / Kp, k = idx - n * Kp;
  float v = (k < K) ? src[k * N + n] : 0.f;
  dst[idx] = (bf16)v;
}

// node features: 4 x embed(64) + 4 scalars, padded to 288, bf16
__global__ __launch_bounds__(256) void embed_nodes_kernel(bf16* __restrict__ out,
    const float* __restrict__ tab, const int* __restrict__ cat,
    const float* __restrict__ scal, int total_gn) {
  int idx = blockIdx.x * blockDim.x + threadIdx.x;
  int gn = idx / DINP_N, col = idx - gn * DINP_N;
  if (gn >= total_gn) return;
  float v = 0.f;
  if (col < 256) {
    int c = col >> 6, e = col & 63;
    int cv = cat[gn * 4 + c];
    v = tab[(c * 16 + cv) * 64 + e];
  } else if (col < 260) {
    v = scal[gn * 4 + (col - 256)];
  }
  out[idx] = (bf16)v;
}

// edge features: 2 x embed(64) + 2 scalars, padded to 160, bf16
__global__ __launch_bounds__(256) void embed_edges_kernel(bf16* __restrict__ out,
    const float* __restrict__ tab, const int* __restrict__ cat,
    const float* __restrict__ scal, int total_ge) {
  int idx = blockIdx.x * blockDim.x + threadIdx.x;
  int ge = idx / DINP_E, col = idx - ge * DINP_E;
  if (ge >= total_ge) return;
  float v = 0.f;
  if (col < 128) {
    int c = col >> 6, e = col & 63;
    int cv = cat[ge * 2 + c];
    v = tab[(c * 8 + cv) * 64 + e];
  } else if (col < 130) {
    v = scal[ge * 2 + (col - 128)];
  }
  out[idx] = (bf16)v;
}

// ------------------------------- WMMA GEMM ---------------------------------
// C[g] = A[g] (MxK, row major bf16) @ B[g]^T (B stored [N,K] row-major bf16)
// + bias, optional ReLU; writes f32 C and/or bf16 shadow Cb.
// One wave computes a 16x64 C tile (4 x v_wmma_f32_16x16x32_bf16 per K-step).
// K is a template parameter: the K-loop fully unrolls into straight-line SSA
// (no phis -> no accumulator/fragment register rotation), letting the
// scheduler hoist the next step's loads above the current WMMAs.

struct GemmCfg {
  const bf16* A; const bf16* B; const float* bias;
  float* C; bf16* Cb;
  long long a_batch, b_batch, c_batch, cb_batch;
  int lda, ldb, ldc, ldcb, M, N, relu;
};

template <int K>
__global__ __launch_bounds__(256) void gemm_bf16_kernel(GemmCfg cfg) {
  const int tid = threadIdx.x;
  const int ntile4 = cfg.N >> 6;                 // N / 64
  const int tiles  = (cfg.M >> 4) * ntile4;
  const int wid    = blockIdx.x * 8 + (tid >> 5);
  if (wid >= tiles) return;                      // uniform per wave: EXEC stays full
  const int mt  = wid / ntile4;
  const int nt4 = wid - mt * ntile4;
  const int g   = blockIdx.z;
  const int lane = tid & 31, half = lane >> 4, l15 = lane & 15;

  const bf16* A = cfg.A + (long long)g * cfg.a_batch;
  const bf16* B = cfg.B + (long long)g * cfg.b_batch;
  // A fragment (16x32 bf16): lanes 0-15 hold K 0..7 / 16..23 of row l15,
  //                          lanes 16-31 hold K 8..15 / 24..31.
  const bf16* arow = A + (long long)(mt * 16 + l15) * cfg.lda + half * 8;
  // B fragment (32x16 bf16): lanes 0-15 hold K 0..7 / 16..23 of column l15,
  //                          lanes 16-31 hold K 8..15 / 24..31.
  const bf16* brow[4];
#pragma unroll
  for (int j = 0; j < 4; ++j)
    brow[j] = B + (long long)((nt4 * 4 + j) * 16 + l15) * cfg.ldb + half * 8;

  v8f acc[4] = {};
#pragma unroll
  for (int k0 = 0; k0 < K; k0 += 32) {
    v8bf alo = *(const v8bf*)(arow + k0);
    v8bf ahi = *(const v8bf*)(arow + k0 + 16);
    v16bf av = __builtin_shufflevector(alo, ahi,
        0,1,2,3,4,5,6,7,8,9,10,11,12,13,14,15);
#pragma unroll
    for (int j = 0; j < 4; ++j) {
      v8bf blo = *(const v8bf*)(brow[j] + k0);
      v8bf bhi = *(const v8bf*)(brow[j] + k0 + 16);
      v16bf bv = __builtin_shufflevector(blo, bhi,
          0,1,2,3,4,5,6,7,8,9,10,11,12,13,14,15);
      acc[j] = __builtin_amdgcn_wmma_f32_16x16x32_bf16(
          false, av, false, bv, (short)0, acc[j], false, false);
    }
  }

  // C layout: VGPR r -> row mt*16 + r + half*8, col = tile*16 + l15
  const int orow = mt * 16 + half * 8;
#pragma unroll
  for (int j = 0; j < 4; ++j) {
    const int col = (nt4 * 4 + j) * 16 + l15;
    const float bv = cfg.bias ? cfg.bias[col] : 0.f;
#pragma unroll
    for (int r = 0; r < 8; ++r) {
      float v = acc[j][r] + bv;
      if (cfg.relu) v = fmaxf(v, 0.f);
      const long long rr = orow + r;
      if (cfg.C)
        cfg.C[(long long)g * cfg.c_batch + rr * cfg.ldc + col] = v;
      if (cfg.Cb)
        cfg.Cb[(long long)g * cfg.cb_batch + rr * cfg.ldcb + col] = (bf16)v;
    }
  }
}

// --------------------------- message passing -------------------------------

__global__ __launch_bounds__(256) void init_agg_kernel(float* __restrict__ agg,
    const float* __restrict__ bconv, int total) {
  int idx = blockIdx.x * blockDim.x + threadIdx.x;
  if (idx >= total) return;
  agg[idx] = bconv[idx & 63];
}

// one wave per edge: msg[e,:] = h[src[e],:] @ W_edge[e]  (bf16 weights),
// atomically accumulated into agg[dst[e],:]
__global__ __launch_bounds__(256) void msg_kernel(float* __restrict__ agg,
    const float* __restrict__ h, const bf16* __restrict__ wedge,
    const int* __restrict__ src, const int* __restrict__ dst, int E, int Nn) {
  const int wid  = (blockIdx.x * 256 + threadIdx.x) >> 5;
  const int lane = threadIdx.x & 31;
  const int g = wid / E, e = wid - g * E;
  if (g >= GG) return;
  const int s = src[g * E + e], d = dst[g * E + e];
  const float* hrow = h + (g * Nn + s) * 64;
  const bf16* w = wedge + (long long)(g * E + e) * 4096 + lane * 2;
  float a0 = 0.f, a1 = 0.f;
#pragma unroll 4
  for (int i = 0; i < 64; ++i) {
    float hv = hrow[i];
    v2bf wv = *(const v2bf*)(w + i * 64);
    a0 += hv * (float)wv[0];
    a1 += hv * (float)wv[1];
  }
  float* ap = agg + (g * Nn + d) * 64 + lane * 2;
  atomicAdd(ap, a0);
  atomicAdd(ap + 1, a1);
}

__global__ __launch_bounds__(256) void relu_bf_kernel(bf16* __restrict__ mbf,
    const float* __restrict__ agg, int total) {
  int idx = blockIdx.x * blockDim.x + threadIdx.x;
  if (idx >= total) return;
  mbf[idx] = (bf16)fmaxf(agg[idx], 0.f);
}

__global__ __launch_bounds__(256) void gru_kernel(float* __restrict__ h,
    bf16* __restrict__ hbf, const float* __restrict__ gi,
    const float* __restrict__ gh, int total) {
  int idx = blockIdx.x * blockDim.x + threadIdx.x;
  if (idx >= total) return;
  int o = idx & 63, gn = idx >> 6;
  const float* gir = gi + gn * 192;
  const float* ghr = gh + gn * 192;
  float r = 1.f / (1.f + expf(-(gir[o] + ghr[o])));
  float z = 1.f / (1.f + expf(-(gir[64 + o] + ghr[64 + o])));
  float n = tanhf(gir[128 + o] + r * ghr[128 + o]);
  float hn = (1.f - z) * n + z * h[idx];
  h[idx] = hn;
  hbf[idx] = (bf16)hn;
}

// ------------------------------- docking -----------------------------------

__device__ __forceinline__ float wave_sum(float v) {
#pragma unroll
  for (int off = 16; off > 0; off >>= 1) v += __shfl_xor(v, off, 32);
  return v;
}

__global__ __launch_bounds__(512) void dock_kernel(float* __restrict__ out,
    const float* __restrict__ atn, const float* __restrict__ ligco,
    const float* __restrict__ recco, const float* __restrict__ rot0,
    const float* __restrict__ trans0) {
  const int g = blockIdx.x, tid = threadIdx.x;
  const int lane = tid & 31, wv = tid >> 5;
  __shared__ float ligc[NLIG][3];
  __shared__ float rot[9], trans[3];
  __shared__ float lmean[3], rmean[3];
  __shared__ float red[16][13];
  __shared__ float acc[13];

  if (tid < 3) {
    float s = 0.f;
    for (int i = 0; i < NLIG; ++i) s += ligco[(g * NLIG + i) * 3 + tid];
    lmean[tid] = s * (1.f / NLIG);
    float t = 0.f;
    for (int j2 = 0; j2 < NREC; ++j2) t += recco[(g * NREC + j2) * 3 + tid];
    rmean[tid] = t * (1.f / NREC);
  }
  if (tid < 9) rot[tid] = rot0[g * 9 + tid];
  if (tid < 3) trans[tid] = trans0[g * 3 + tid];
  __syncthreads();
  if (tid < NLIG) {
    ligc[tid][0] = ligco[(g * NLIG + tid) * 3 + 0] - lmean[0];
    ligc[tid][1] = ligco[(g * NLIG + tid) * 3 + 1] - lmean[1];
    ligc[tid][2] = ligco[(g * NLIG + tid) * 3 + 2] - lmean[2];
  }
  const int j = tid;  // this thread owns receptor atom j
  const float rjx = recco[(g * NREC + j) * 3 + 0] - rmean[0];
  const float rjy = recco[(g * NREC + j) * 3 + 1] - rmean[1];
  const float rjz = recco[(g * NREC + j) * 3 + 2] - rmean[2];
  __syncthreads();

  const float scale = 1.f / (float)(NLIG * NREC);
  float Uout = 0.f;
  for (int k = 0; k < 8; ++k) {
    float R0 = rot[0], R1 = rot[1], R2 = rot[2];
    float R3 = rot[3], R4 = rot[4], R5 = rot[5];
    float R6 = rot[6], R7 = rot[7], R8 = rot[8];
    float T0 = trans[0], T1 = trans[1], T2 = trans[2];
    float v[13];
#pragma unroll
    for (int t = 0; t < 13; ++t) v[t] = 0.f;
    for (int i = 0; i < NLIG; ++i) {
      float lx = ligc[i][0], ly = ligc[i][1], lz = ligc[i][2];
      float nx = R0 * lx + R1 * ly + R2 * lz + T0;
      float ny = R3 * lx + R4 * ly + R5 * lz + T1;
      float nz = R6 * lx + R7 * ly + R8 * lz + T2;
      float dx = nx - rjx, dy = ny - rjy, dz = nz - rjz;
      float d2 = dx * dx + dy * dy + dz * dz;
      float a = atn[(g * NLIG + i) * NREC + j];
      float inv = 1.f / d2;
      float w = a * inv;      // atn/d2       -> energy
      float c2 = w * inv;     // atn/d2^2     -> gradient weight
      v[0]  += w;
      v[1]  += c2 * dx;  v[2]  += c2 * dy;  v[3]  += c2 * dz;
      v[4]  += c2 * dx * lx; v[5]  += c2 * dx * ly; v[6]  += c2 * dx * lz;
      v[7]  += c2 * dy * lx; v[8]  += c2 * dy * ly; v[9]  += c2 * dy * lz;
      v[10] += c2 * dz * lx; v[11] += c2 * dz * ly; v[12] += c2 * dz * lz;
    }
#pragma unroll
    for (int t = 0; t < 13; ++t) {
      float s = wave_sum(v[t]);
      if (lane == 0) red[wv][t] = s;
    }
    __syncthreads();
    if (tid < 13) {
      float s = 0.f;
      for (int u = 0; u < 16; ++u) s += red[u][tid];
      acc[tid] = s;
    }
    __syncthreads();
    if (k == 7) {
      Uout = scale * acc[0];                       // Us[-1]: after 7 updates
    } else {
      // grad = -2*scale*acc  ->  p -= lr*grad  ==  p += 2*scale*lr*acc
      if (tid < 3) trans[tid] += (2.f * scale * OPT_LRF) * acc[1 + tid];
      if (tid < 9) rot[tid]   += (2.f * scale * OPT_LRF) * acc[4 + tid];
    }
    __syncthreads();
  }
  if (tid == 0) out[g] = -Uout;
}

// ------------------------------- host side ---------------------------------

extern "C" void kernel_launch(void* const* d_in, const int* in_sizes, int n_in,
                              void* d_out, int out_size, void* d_ws, size_t ws_size,
                              hipStream_t stream) {
  (void)out_size; (void)ws_size;
  // Logical input slots (insertion order of setup_inputs / nested params):
  //  0..3  lig_node_tab, rec_node_tab, lig_edge_tab, rec_edge_tab
  //  4..14 lig gnn: W_proj,b_proj,W_e1,b_e1,W_e2,b_e2,b_conv,W_ih,b_ih,W_hh,b_hh
  // 15..25 rec gnn (same order)
  // 26..32 lig_cat, lig_scal, lig_coord, lig_src, lig_dst, lig_e_cat, lig_e_scal
  // 33..39 rec_cat, rec_scal, rec_coord, rec_src, rec_dst, rec_e_cat, rec_e_scal
  // 40,41  rot_init, trans_init
  const void* P[42];
  if (n_in == 17) {
    // params passed as one concatenated blob
    const float* blob = (const float*)d_in[0];
    static const long long boff[26] = {
      0, 4096, 8192, 9216,
      10240, 26880, 26944, 43584, 43712, 568000, 572096, 572160, 584448, 584640, 596928,
      597120, 613760, 613824, 630464, 630592, 1154880, 1158976, 1159040, 1171328, 1171520, 1183808
    };
    for (int i = 0; i < 26; ++i) P[i] = blob + boff[i];
    for (int i = 0; i < 16; ++i) P[26 + i] = d_in[1 + i];
  } else if (n_in >= 42 && in_sizes[0] == 3072) {
    // jax pytree (sorted-key) leaf order
    static const int smap[42] = {
      19, 32, 18, 31,
      11, 17, 7, 13, 8, 14, 12, 10, 16, 9, 15,
      24, 30, 20, 26, 21, 27, 25, 23, 29, 22, 28,
      0, 5, 1, 6, 2, 3, 4,
      33, 38, 34, 39, 35, 36, 37,
      40, 41};
    for (int i = 0; i < 42; ++i) P[i] = d_in[smap[i]];
  } else {
    for (int i = 0; i < 42 && i < n_in; ++i) P[i] = d_in[i];
  }

  struct TB {
    const float *Wproj, *bproj, *We1, *be1, *We2, *be2, *bconv, *Wih, *bih, *Whh, *bhh;
    const int *cat, *srcI, *dstI, *ecat;
    const float *scal, *escal;
    int Nn, Ne;
    bf16 *wprojT, *we1T, *we2T, *wihT, *whhT;
    bf16 *nodefeat, *edgefeat, *eh, *wedge, *hbf, *mbf;
    float *h, *agg, *gi, *gh;
  } tb[2];

  const float* ntab[2] = {(const float*)P[0], (const float*)P[1]};
  const float* etab[2] = {(const float*)P[2], (const float*)P[3]};
  for (int t = 0; t < 2; ++t) {
    int b = 4 + t * 11;
    tb[t].Wproj = (const float*)P[b + 0];  tb[t].bproj = (const float*)P[b + 1];
    tb[t].We1   = (const float*)P[b + 2];  tb[t].be1   = (const float*)P[b + 3];
    tb[t].We2   = (const float*)P[b + 4];  tb[t].be2   = (const float*)P[b + 5];
    tb[t].bconv = (const float*)P[b + 6];
    tb[t].Wih   = (const float*)P[b + 7];  tb[t].bih   = (const float*)P[b + 8];
    tb[t].Whh   = (const float*)P[b + 9];  tb[t].bhh   = (const float*)P[b + 10];
  }
  tb[0].cat = (const int*)P[26];   tb[0].scal = (const float*)P[27];
  const float* lig_coord = (const float*)P[28];
  tb[0].srcI = (const int*)P[29];  tb[0].dstI = (const int*)P[30];
  tb[0].ecat = (const int*)P[31];  tb[0].escal = (const float*)P[32];
  tb[1].cat = (const int*)P[33];   tb[1].scal = (const float*)P[34];
  const float* rec_coord = (const float*)P[35];
  tb[1].srcI = (const int*)P[36];  tb[1].dstI = (const int*)P[37];
  tb[1].ecat = (const int*)P[38];  tb[1].escal = (const float*)P[39];
  const float* rot0 = (const float*)P[40];
  const float* trans0 = (const float*)P[41];
  tb[0].Nn = NLIG; tb[0].Ne = ELIG;
  tb[1].Nn = NREC; tb[1].Ne = EREC;

  // workspace carve-up (256B aligned); W_edge buffers (151 MB bf16) go last
  char* base = (char*)d_ws;
  size_t off = 0;
  auto alloc = [&](size_t bytes) -> void* {
    void* p = base + off;
    off += (bytes + 255) & ~(size_t)255;
    return p;
  };
  for (int t = 0; t < 2; ++t) {
    TB& T = tb[t];
    T.wprojT   = (bf16*)alloc((size_t)64 * DINP_N * 2);
    T.we1T     = (bf16*)alloc((size_t)128 * DINP_E * 2);
    T.we2T     = (bf16*)alloc((size_t)4096 * 128 * 2);
    T.wihT     = (bf16*)alloc((size_t)192 * 64 * 2);
    T.whhT     = (bf16*)alloc((size_t)192 * 64 * 2);
    T.nodefeat = (bf16*)alloc((size_t)GG * T.Nn * DINP_N * 2);
    T.edgefeat = (bf16*)alloc((size_t)GG * T.Ne * DINP_E * 2);
    T.eh       = (bf16*)alloc((size_t)GG * T.Ne * 128 * 2);
    T.hbf      = (bf16*)alloc((size_t)GG * T.Nn * 64 * 2);
    T.mbf      = (bf16*)alloc((size_t)GG * T.Nn * 64 * 2);
    T.h        = (float*)alloc((size_t)GG * T.Nn * 64 * 4);
    T.agg      = (float*)alloc((size_t)GG * T.Nn * 64 * 4);
    T.gi       = (float*)alloc((size_t)GG * T.Nn * 192 * 4);
    T.gh       = (float*)alloc((size_t)GG * T.Nn * 192 * 4);
  }
  float* atn = (float*)alloc((size_t)GG * NLIG * NREC * 4);
  for (int t = 0; t < 2; ++t)
    tb[t].wedge = (bf16*)alloc((size_t)GG * tb[t].Ne * 4096 * 2);

  auto grid1 = [](size_t n) { return dim3((unsigned)((n + 255) / 256)); };
  auto gemm = [&](const bf16* A, long long ab, int lda,
                  const bf16* B, long long bb, int ldb, const float* bias,
                  float* C, long long cb, int ldc,
                  bf16* Cb, long long cbb, int ldcb,
                  int M, int N, int K, int relu) {
    GemmCfg cfg;
    cfg.A = A; cfg.B = B; cfg.bias = bias; cfg.C = C; cfg.Cb = Cb;
    cfg.a_batch = ab; cfg.b_batch = bb; cfg.c_batch = cb; cfg.cb_batch = cbb;
    cfg.lda = lda; cfg.ldb = ldb; cfg.ldc = ldc; cfg.ldcb = ldcb;
    cfg.M = M; cfg.N = N; cfg.relu = relu;
    int tiles = (M / 16) * (N / 64);
    dim3 grid((tiles + 7) / 8, 1, GG);
    switch (K) {
      case DINP_N: gemm_bf16_kernel<DINP_N><<<grid, 256, 0, stream>>>(cfg); break;
      case DINP_E: gemm_bf16_kernel<DINP_E><<<grid, 256, 0, stream>>>(cfg); break;
      case 128:    gemm_bf16_kernel<128><<<grid, 256, 0, stream>>>(cfg);    break;
      default:     gemm_bf16_kernel<64><<<grid, 256, 0, stream>>>(cfg);     break;
    }
  };

  for (int t = 0; t < 2; ++t) {
    TB& T = tb[t];
    // weights -> transposed bf16 ([N, Kpad])
    wtrans_kernel<<<grid1((size_t)64 * DINP_N), 256, 0, stream>>>(T.wprojT, T.Wproj, 260, 64, DINP_N);
    wtrans_kernel<<<grid1((size_t)128 * DINP_E), 256, 0, stream>>>(T.we1T, T.We1, 130, 128, DINP_E);
    wtrans_kernel<<<grid1((size_t)4096 * 128), 256, 0, stream>>>(T.we2T, T.We2, 128, 4096, 128);
    wtrans_kernel<<<grid1((size_t)192 * 64), 256, 0, stream>>>(T.wihT, T.Wih, 64, 192, 64);
    wtrans_kernel<<<grid1((size_t)192 * 64), 256, 0, stream>>>(T.whhT, T.Whh, 64, 192, 64);
    // embeddings -> padded bf16 features
    embed_nodes_kernel<<<grid1((size_t)GG * T.Nn * DINP_N), 256, 0, stream>>>(
        T.nodefeat, ntab[t], T.cat, T.scal, GG * T.Nn);
    embed_edges_kernel<<<grid1((size_t)GG * T.Ne * DINP_E), 256, 0, stream>>>(
        T.edgefeat, etab[t], T.ecat, T.escal, GG * T.Ne);
    // h0 = relu(nodefeat @ W_proj + b)
    gemm(T.nodefeat, (long long)T.Nn * DINP_N, DINP_N, T.wprojT, 0, DINP_N,
         T.bproj, T.h, (long long)T.Nn * 64, 64, T.hbf, (long long)T.Nn * 64, 64,
         T.Nn, 64, DINP_N, 1);
    // eh = relu(edgefeat @ W_e1 + b)            (bf16 only)
    gemm(T.edgefeat, (long long)T.Ne * DINP_E, DINP_E, T.we1T, 0, DINP_E,
         T.be1, nullptr, 0, 0, T.eh, (long long)T.Ne * 128, 128,
         T.Ne, 128, DINP_E, 1);
    // W_edge = eh @ W_e2 + b                    (dominant GEMM; bf16 store)
    gemm(T.eh, (long long)T.Ne * 128, 128, T.we2T, 0, 128,
         T.be2, nullptr, 0, 0, T.wedge, (long long)T.Ne * 4096, 4096,
         T.Ne, 4096, 128, 0);
    // message passing x4
    const int tot = GG * T.Nn * 64;
    for (int s = 0; s < 4; ++s) {
      init_agg_kernel<<<grid1((size_t)tot), 256, 0, stream>>>(T.agg, T.bconv, tot);
      msg_kernel<<<dim3(GG * T.Ne / 8), 256, 0, stream>>>(
          T.agg, T.h, T.wedge, T.srcI, T.dstI, T.Ne, T.Nn);
      relu_bf_kernel<<<grid1((size_t)tot), 256, 0, stream>>>(T.mbf, T.agg, tot);
      gemm(T.mbf, (long long)T.Nn * 64, 64, T.wihT, 0, 64, T.bih,
           T.gi, (long long)T.Nn * 192, 192, nullptr, 0, 0, T.Nn, 192, 64, 0);
      gemm(T.hbf, (long long)T.Nn * 64, 64, T.whhT, 0, 64, T.bhh,
           T.gh, (long long)T.Nn * 192, 192, nullptr, 0, 0, T.Nn, 192, 64, 0);
      gru_kernel<<<grid1((size_t)tot), 256, 0, stream>>>(T.h, T.hbf, T.gi, T.gh, tot);
    }
  }

  // atn[g] = lig_feat[g] @ rec_feat[g]^T  (per-graph B via b_batch)
  gemm(tb[0].hbf, (long long)NLIG * 64, 64, tb[1].hbf, (long long)NREC * 64, 64,
       nullptr, atn, (long long)NLIG * NREC, NREC, nullptr, 0, 0,
       NLIG, NREC, 64, 0);

  // 8-step rigid docking optimization, one block per graph
  dock_kernel<<<dim3(GG), 512, 0, stream>>>((float*)d_out, atn, lig_coord,
                                            rec_coord, rot0, trans0);
}
